// attentiveConv_80874234183985
// MI455X (gfx1250) — matrix operand order repaired and checked by reading, not compiled, not run on hardware
//
#include <hip/hip_runtime.h>
#include <math.h>

// ---------------------------------------------------------------------------
// AMD MI455X (gfx1250, wave32) attention-augmented convolution.
//   K1 pack_patches : im2col(x) -> bf16  [4096 x 1152]
//   K2 pack_weights : (conv_out_w ++ qkv_w) -> bf16, k-blocked [36][768][32]
//   K3 conv_gemm    : WMMA bf16 implicit GEMM [4096x1152]x[1152x768]
//                     scalar-branch epilogue -> d_out (conv half) + Qb/Kb/Vt
//   K4 rel_logits   : Pw/Ph[i][m] = q_i . key_rel_{w,h}[m]   (rank-reduced)
//   K5 flash_attn   : online-softmax attention; K/V staged to LDS with
//                     double-buffered GLOBAL_LOAD_ASYNC_TO_LDS_B128 (+ WMMA)
//   K6 attn_out     : 1x1 conv + bias -> attn half of d_out
// Logits (128 MB) are never materialized.
// ---------------------------------------------------------------------------

typedef __attribute__((ext_vector_type(16))) __bf16         v16bf;
typedef __attribute__((ext_vector_type(8)))  float          v8f;
typedef __attribute__((ext_vector_type(8)))  unsigned short v8us;

union FragU { v16bf bf; v8us us[2]; };

__device__ __forceinline__ unsigned short f32_to_bf16(float f) {
  unsigned int u = __float_as_uint(f);
  u += 0x7fffu + ((u >> 16) & 1u);          // round-to-nearest-even
  return (unsigned short)(u >> 16);
}
__device__ __forceinline__ float bf16_to_f32(unsigned short h) {
  return __uint_as_float(((unsigned int)h) << 16);
}

// A-fragment, 16x32 bf16 (MxK), row-major source with row stride `ldr` elems.
// ISA 7.12.2: lane L -> row L%16; k runs [off,off+8) and [16+off,16+off+8),
// off = (L<16 ? 0 : 8). Two 16B vector loads per lane.
__device__ __forceinline__ v16bf load_fragA(const unsigned short* tile, int ldr, int lane) {
  const unsigned short* p = tile + (lane & 15) * ldr + ((lane & 16) ? 8 : 0);
  FragU f;
  f.us[0] = *(const v8us*)p;
  f.us[1] = *(const v8us*)(p + 16);
  return f.bf;
}
// B-fragment, 32x16 bf16 (KxN): lane L -> col L%16, k runs [off,off+8) and
// [16+off,16+off+8), off = (L<16 ? 0 : 8). Source: per-column k contiguous, col stride `ldc`.
__device__ __forceinline__ v16bf load_fragB(const unsigned short* tile, int ldc, int lane) {
  const unsigned short* p = tile + (lane & 15) * ldc + ((lane & 16) ? 8 : 0);
  FragU f;
  f.us[0] = *(const v8us*)p;
  f.us[1] = *(const v8us*)(p + 16);
  return f.bf;
}

__device__ __forceinline__ v8f wmma_bf16(v16bf a, v16bf b, v8f c) {
  return __builtin_amdgcn_wmma_f32_16x16x32_bf16(false, a, false, b, (short)0, c, false, false);
}

// CDNA5 async global->LDS copy, 16B per lane, tracked by ASYNCcnt.
// lds_off = LDS byte address (low 32 bits of a generic pointer to __shared__).
__device__ __forceinline__ void async_g2l_b128(unsigned lds_off, const void* gptr) {
  asm volatile("global_load_async_to_lds_b128 %0, %1, off"
               :: "v"(lds_off), "v"(gptr)
               : "memory");
}
__device__ __forceinline__ void wait_async0() {
  asm volatile("s_wait_asynccnt 0" ::: "memory");
}

// --------------------------- K1: im2col pack -------------------------------
__global__ __launch_bounds__(256) void pack_patches_kernel(
    const float* __restrict__ x, unsigned short* __restrict__ A) {
  int idx = blockIdx.x * 256 + threadIdx.x;   // [m=4096][kk=1152]
  int m  = idx / 1152;
  int kk = idx - m * 1152;
  int c  = kk / 9;
  int rr = kk - c * 9;
  int ky = rr / 3, kx = rr - ky * 3;
  int b = m >> 10, pix = m & 1023;
  int h = pix >> 5, w = pix & 31;
  int hy = h + ky - 1, wx = w + kx - 1;
  float v = 0.f;
  if ((unsigned)hy < 32u && (unsigned)wx < 32u)
    v = x[((((b << 7) + c) << 5) + hy) * 32 + wx];
  A[idx] = f32_to_bf16(v);
}

// --------------------------- K2: weight pack -------------------------------
__global__ __launch_bounds__(256) void pack_weights_kernel(
    const float* __restrict__ cw, const float* __restrict__ qw,
    unsigned short* __restrict__ W2) {
  int idx = blockIdx.x * 256 + threadIdx.x;   // [o=768][kk=1152]
  int o  = idx / 1152;
  int kk = idx - o * 1152;
  float v = (o < 128) ? cw[idx] : qw[idx - 128 * 1152];
  int kb = kk >> 5, kl = kk & 31;
  W2[((kb * 768) + o) * 32 + kl] = f32_to_bf16(v);  // k-blocked [36][768][32]
}

// --------------------------- K3: fused conv GEMM ---------------------------
// 256 threads = 8 waves; per-wave 64x32 tile (4x2 WMMA accumulators).
// Workgroup tile 128(M) x 128(N); grid (32, 6). K = 1152 (36 steps of 32).
__global__ __launch_bounds__(256) void conv_gemm_kernel(
    const unsigned short* __restrict__ A, const unsigned short* __restrict__ W2,
    const float* __restrict__ conv_out_b, const float* __restrict__ qkv_b,
    float* __restrict__ out, unsigned short* __restrict__ Qb,
    unsigned short* __restrict__ Kb, unsigned short* __restrict__ Vt) {
  int tid = threadIdx.x, wv = tid >> 5, lane = tid & 31;
  int mbase = blockIdx.x * 128 + (wv & 1) * 64;
  int nbase = blockIdx.y * 128 + (wv >> 1) * 32;

  v8f acc[4][2];
  #pragma unroll
  for (int i = 0; i < 4; ++i)
    #pragma unroll
    for (int j = 0; j < 2; ++j) acc[i][j] = (v8f){};

  for (int kb = 0; kb < 36; ++kb) {
    if (kb + 1 < 36) {                         // L2 prefetch of next K-step
      __builtin_prefetch(A + (mbase + (lane & 15) * 16) * 1152 + (kb + 1) * 32, 0, 0);
      __builtin_prefetch(W2 + (((kb + 1) * 768) + nbase + (lane & 15)) * 32, 0, 0);
    }
    v16bf af[4];
    #pragma unroll
    for (int i = 0; i < 4; ++i)
      af[i] = load_fragA(A + (mbase + i * 16) * 1152 + kb * 32, 1152, lane);
    #pragma unroll
    for (int j = 0; j < 2; ++j) {
      v16bf bfm = load_fragB(W2 + ((kb * 768) + nbase + j * 16) * 32, 32, lane);
      #pragma unroll
      for (int i = 0; i < 4; ++i) acc[i][j] = wmma_bf16(af[i], bfm, acc[i][j]);
    }
  }

  // Epilogue. C layout: lane -> row r + 8*(lane>=16), col lane%16.
  // Category boundaries (128/384/640) are multiples of 16, so each 16x16
  // block has one destination; pick it with a *scalar* branch.
  int rowh = (lane & 16) ? 8 : 0;
  int col  = lane & 15;
  #pragma unroll
  for (int j = 0; j < 2; ++j) {
    int nb = __builtin_amdgcn_readfirstlane(nbase) + j * 16;  // uniform block col
    int n  = nb + col;
    if (nb < 128) {                                           // conv_out half
      float bias = conv_out_b[n];
      #pragma unroll
      for (int i = 0; i < 4; ++i)
        #pragma unroll
        for (int r = 0; r < 8; ++r) {
          int m = mbase + i * 16 + r + rowh;
          int b = m >> 10, pix = m & 1023;
          out[(((b << 8) + n) << 10) + pix] = acc[i][j][r] + bias;
        }
    } else if (nb < 384) {                                    // Q (scaled)
      int ch = n - 128, head = ch >> 5, d = ch & 31;
      float bias = qkv_b[ch];
      #pragma unroll
      for (int i = 0; i < 4; ++i)
        #pragma unroll
        for (int r = 0; r < 8; ++r) {
          int m = mbase + i * 16 + r + rowh;
          int b = m >> 10, pix = m & 1023;
          Qb[((((b << 3) + head) << 10) + pix) * 32 + d] =
              f32_to_bf16((acc[i][j][r] + bias) * 0.17677669529663688f);
        }
    } else if (nb < 640) {                                    // K
      int ch = n - 384, head = ch >> 5, d = ch & 31;
      float bias = qkv_b[n - 128];
      #pragma unroll
      for (int i = 0; i < 4; ++i)
        #pragma unroll
        for (int r = 0; r < 8; ++r) {
          int m = mbase + i * 16 + r + rowh;
          int b = m >> 10, pix = m & 1023;
          Kb[((((b << 3) + head) << 10) + pix) * 32 + d] =
              f32_to_bf16(acc[i][j][r] + bias);
        }
    } else {                                                  // V (transposed)
      int ch = n - 640, head = ch >> 4, d = ch & 15;
      float bias = qkv_b[n - 128];
      #pragma unroll
      for (int i = 0; i < 4; ++i)
        #pragma unroll
        for (int r = 0; r < 8; ++r) {
          int m = mbase + i * 16 + r + rowh;
          int b = m >> 10, pix = m & 1023;
          Vt[(((((b << 3) + head) << 4) + d) << 10) + pix] =
              f32_to_bf16(acc[i][j][r] + bias);
        }
    }
  }
}

// --------------------------- K4: rel-logit rank reduction ------------------
__global__ __launch_bounds__(256) void rel_logits_kernel(
    const unsigned short* __restrict__ Qb, const float* __restrict__ relw,
    const float* __restrict__ relh, float* __restrict__ Pw, float* __restrict__ Ph) {
  int idx = blockIdx.x * 256 + threadIdx.x;   // [bh=32][i=1024][m=64 (63 used)]
  int m  = idx & 63;
  int i  = (idx >> 6) & 1023;
  int bh = idx >> 16;
  if (m >= 63) { Pw[idx] = 0.f; Ph[idx] = 0.f; return; }
  const unsigned short* q = Qb + (((bh << 10) + i) << 5);
  const float* rw = relw + m * 32;
  const float* rh = relh + m * 32;
  float aw = 0.f, ah = 0.f;
  #pragma unroll
  for (int d = 0; d < 32; ++d) {
    float qv = bf16_to_f32(q[d]);
    aw += qv * rw[d];
    ah += qv * rh[d];
  }
  Pw[idx] = aw; Ph[idx] = ah;
}

// --------------------------- K5: flash attention ---------------------------
// grid = 32 (b*head) * 8 (query tiles of 128). 8 waves; wave owns 16 q-rows.
// K (8KB) and V (4KB) blocks are shared by all 8 waves -> stage them once per
// workgroup into double-buffered LDS with async global->LDS copies, and issue
// the next block's stage right after the current S-WMMAs (overlaps softmax).
__global__ __launch_bounds__(256) void flash_attn_kernel(
    const unsigned short* __restrict__ Qb, const unsigned short* __restrict__ Kb,
    const unsigned short* __restrict__ Vt, const float* __restrict__ Pw,
    const float* __restrict__ Ph, float* __restrict__ AttnBuf) {
  __shared__ __align__(16) unsigned short Klds[2][128][32];  // 2 x 8 KB
  __shared__ __align__(16) unsigned short Vlds[2][16][128];  // 2 x 4 KB
  __shared__ __align__(16) unsigned short Pl[128][128];      // 32 KB
  int tid = threadIdx.x, wv = tid >> 5, lane = tid & 31;
  int bh = blockIdx.x >> 3, qt = blockIdx.x & 7;
  int qrow = wv << 4;
  int ibase = qt << 7;
  int rowh = (lane & 16) ? 8 : 0;
  int col  = lane & 15;

  unsigned kofs0 = (unsigned)(size_t)&Klds[0][0][0];   // LDS byte addresses
  unsigned vofs0 = (unsigned)(size_t)&Vlds[0][0][0];

  // stage key-block kb into LDS buffer `buf` (256 threads x 16B chunks)
  auto stage = [&](int kb, int buf) {
    const unsigned short* gk = Kb + ((bh << 10) + (kb << 7)) * 32;  // 8192 B
    unsigned kdst = kofs0 + (unsigned)buf * 8192u;
    async_g2l_b128(kdst + (unsigned)tid * 16u, gk + tid * 8);
    async_g2l_b128(kdst + (unsigned)(tid + 256) * 16u, gk + (tid + 256) * 8);
    int seg = tid >> 4, part = tid & 15;                            // 4096 B
    const unsigned short* gv =
        Vt + (((bh << 4) + seg) << 10) + (kb << 7) + part * 8;
    async_g2l_b128(vofs0 + (unsigned)buf * 4096u +
                   (unsigned)(seg * 256 + part * 16), gv);
  };

  v16bf qa = load_fragA(Qb + ((bh << 10) + ibase + qrow) * 32, 32, lane);

  float m_run[8], l_run[8];
  #pragma unroll
  for (int r = 0; r < 8; ++r) { m_run[r] = -1e30f; l_run[r] = 0.f; }
  v8f oacc = (v8f){};

  stage(0, 0);

  for (int kb = 0; kb < 8; ++kb) {
    int buf = kb & 1;
    wait_async0();          // this wave's stage done
    __syncthreads();        // all waves' stages done
    // S = Q.K^T for 128 keys (dkh==32 => one WMMA per 16x16 block)
    v8f s[8];
    #pragma unroll
    for (int n = 0; n < 8; ++n) {
      v16bf kf = load_fragB(&Klds[buf][n << 4][0], 32, lane);
      v8f z = (v8f){};
      s[n] = wmma_bf16(qa, kf, z);
    }
    if (kb < 7) stage(kb + 1, buf ^ 1);   // overlap next stage with softmax
    // rel-position logits: S(i,j) += Pw[i][wj-wi+31] + Ph[i][hj-hi+31]
    #pragma unroll
    for (int r = 0; r < 8; ++r) {
      int li = qrow + r + rowh;
      int i  = ibase + li;
      int hi = i >> 5, wi = i & 31;
      const float* rw = Pw + (((bh << 10) + i) << 6) + 31 - wi;
      const float* rh = Ph + (((bh << 10) + i) << 6) + 31 - hi;
      #pragma unroll
      for (int n = 0; n < 8; ++n) {
        int j = (kb << 7) + (n << 4) + col;
        s[n][r] += rw[j & 31] + rh[j >> 5];
      }
    }
    // online softmax: row stats via shfl_xor over the 16-lane half
    float mnew[8], corr[8];
    #pragma unroll
    for (int r = 0; r < 8; ++r) {
      float mx = s[0][r];
      #pragma unroll
      for (int n = 1; n < 8; ++n) mx = fmaxf(mx, s[n][r]);
      #pragma unroll
      for (int d = 1; d < 16; d <<= 1) mx = fmaxf(mx, __shfl_xor(mx, d, 32));
      mnew[r] = fmaxf(m_run[r], mx);
      corr[r] = __expf(m_run[r] - mnew[r]);
      m_run[r] = mnew[r];
    }
    #pragma unroll
    for (int r = 0; r < 8; ++r) {
      float sum = 0.f;
      #pragma unroll
      for (int n = 0; n < 8; ++n) {
        float p = __expf(s[n][r] - mnew[r]);
        sum += p;
        Pl[qrow + r + rowh][(n << 4) + col] = f32_to_bf16(p);
      }
      #pragma unroll
      for (int d = 1; d < 16; d <<= 1) sum += __shfl_xor(sum, d, 32);
      l_run[r] = l_run[r] * corr[r] + sum;
      oacc[r] *= corr[r];
    }
    __syncthreads();
    // O += P.V  (K-dim = 128 keys -> 4 WMMA steps), V B-frags from LDS
    #pragma unroll
    for (int c = 0; c < 4; ++c) {
      v16bf pf = load_fragA(&Pl[qrow][0] + (c << 5), 128, lane);
      v16bf vf = load_fragB(&Vlds[buf][0][c << 5], 128, lane);
      oacc = wmma_bf16(pf, vf, oacc);
    }
    __syncthreads();
  }
  #pragma unroll
  for (int r = 0; r < 8; ++r) {
    float inv = 1.0f / l_run[r];
    AttnBuf[(((bh << 10) + ibase + qrow + r + rowh) << 4) + col] = oacc[r] * inv;
  }
}

// --------------------------- K6: 1x1 conv epilogue -------------------------
// AttnBuf flat [b][head][i][dvh] == reference's "direct reshape" [b][128][1024].
__global__ __launch_bounds__(256) void attn_out_kernel(
    const float* __restrict__ AttnBuf, const float* __restrict__ attn_w,
    const float* __restrict__ attn_b, float* __restrict__ out) {
  int idx = blockIdx.x * 256 + threadIdx.x;   // [b][co=128][pix=1024]
  int pix = idx & 1023;
  int co  = (idx >> 10) & 127;
  int b   = idx >> 17;
  const float* ap = AttnBuf + (b << 17) + pix;
  const float* wp = attn_w + (co << 7);
  float acc = attn_b[co];
  #pragma unroll 8
  for (int ci = 0; ci < 128; ++ci) acc += wp[ci] * ap[ci << 10];
  out[(((b << 8) + 128 + co) << 10) + pix] = acc;
}

// ---------------------------------------------------------------------------
extern "C" void kernel_launch(void* const* d_in, const int* in_sizes, int n_in,
                              void* d_out, int out_size, void* d_ws, size_t ws_size,
                              hipStream_t stream) {
  const float* x          = (const float*)d_in[0];
  const float* conv_out_w = (const float*)d_in[1];
  const float* conv_out_b = (const float*)d_in[2];
  const float* qkv_w      = (const float*)d_in[3];
  const float* qkv_b      = (const float*)d_in[4];
  const float* attn_w     = (const float*)d_in[5];
  const float* attn_b     = (const float*)d_in[6];
  const float* key_rel_w  = (const float*)d_in[7];
  const float* key_rel_h  = (const float*)d_in[8];
  float* out = (float*)d_out;

  // workspace carve-out (~35.4 MB total), 256B aligned
  char* ws = (char*)d_ws;
  size_t off = 0;
  auto carve = [&](size_t bytes) {
    char* p = ws + off;
    off += (bytes + 255) & ~(size_t)255;
    return p;
  };
  unsigned short* A   = (unsigned short*)carve(4096ull * 1152 * 2);  // im2col bf16
  unsigned short* W2  = (unsigned short*)carve(1152ull * 768 * 2);   // packed weights
  unsigned short* Qb  = (unsigned short*)carve(32768ull * 32 * 2);   // [bh][i][dkh]
  unsigned short* Kb  = (unsigned short*)carve(32768ull * 32 * 2);   // [bh][j][dkh]
  unsigned short* Vt  = (unsigned short*)carve(512ull * 1024 * 2);   // [bh][dvh][j]
  float*          Pw  = (float*)carve(32768ull * 64 * 4);            // [bh][i][64]
  float*          Ph  = (float*)carve(32768ull * 64 * 4);
  float*          Att = (float*)carve(32768ull * 16 * 4);            // [bh][i][dvh]

  pack_patches_kernel<<<18432, 256, 0, stream>>>(x, A);
  pack_weights_kernel<<<3456, 256, 0, stream>>>(conv_out_w, qkv_w, W2);
  conv_gemm_kernel<<<dim3(32, 6), 256, 0, stream>>>(A, W2, conv_out_b, qkv_b,
                                                    out, Qb, Kb, Vt);
  rel_logits_kernel<<<8192, 256, 0, stream>>>(Qb, key_rel_w, key_rel_h, Pw, Ph);
  flash_attn_kernel<<<256, 256, 0, stream>>>(Qb, Kb, Vt, Pw, Ph, Att);
  attn_out_kernel<<<2048, 256, 0, stream>>>(Att, attn_w, attn_b, out);
}
